// AttentionLSTM_24867860644473
// MI455X (gfx1250) — matrix operand order repaired and checked by reading, not compiled, not run on hardware
//
#include <hip/hip_runtime.h>

// Problem constants (from reference): N=256 batch, T=64 steps, D=H=1024.
#define NB 256
#define TT 64
#define DD 1024
#define HH 1024
#define KK 3072          // D + H + H  (concat [xt | h | attn])
#define NC 4096          // 4*H gate columns

typedef __attribute__((ext_vector_type(16))) __bf16         v16bf;
typedef __attribute__((ext_vector_type(8)))  float          v8f;
typedef __attribute__((ext_vector_type(8)))  unsigned short v8us;

union FragBF {
  v16bf v;
  v8us  h[2];
};

static __device__ __forceinline__ unsigned short f2bf(float f) {
  // round-to-nearest-even fp32 -> bf16
  unsigned int u = __float_as_uint(f);
  unsigned int r = u + 0x7FFFu + ((u >> 16) & 1u);
  return (unsigned short)(r >> 16);
}

static __device__ __forceinline__ float sigf(float x) {
  return 1.0f / (1.0f + __expf(-x));
}

// ---------------------------------------------------------------------------
// Build WT[4096][3072] bf16 = transpose of [Wx; Wh; Wattn] (each 1024x4096).
// Column-major-per-output-channel so WMMA B fragments load contiguously.
// ---------------------------------------------------------------------------
__global__ void convert_weights(const float* __restrict__ Wx,
                                const float* __restrict__ Wh,
                                const float* __restrict__ Wa,
                                unsigned short* __restrict__ WT) {
  long idx = (long)blockIdx.x * blockDim.x + threadIdx.x;  // j*KK + k
  if (idx >= (long)NC * KK) return;
  int j = (int)(idx / KK);   // output gate column 0..4095
  int k = (int)(idx % KK);   // concat-K index 0..3071
  float v;
  if (k < DD)            v = Wx[(long)k * NC + j];
  else if (k < DD + HH)  v = Wh[(long)(k - DD) * NC + j];
  else                   v = Wa[(long)(k - DD - HH) * NC + j];
  WT[idx] = f2bf(v);
}

__global__ void convert_f32_bf16(const float* __restrict__ src,
                                 unsigned short* __restrict__ dst, long n) {
  long i = (long)blockIdx.x * blockDim.x + threadIdx.x;
  if (i < n) dst[i] = f2bf(src[i]);
}

// h0 = c0 = mean over the 16 spatial positions of A[n][h][4][4]
__global__ void init_state(const float* __restrict__ A,
                           float* __restrict__ hf,
                           unsigned short* __restrict__ hb,
                           float* __restrict__ c) {
  int i = blockIdx.x * blockDim.x + threadIdx.x;  // over N*H
  const float* a = A + ((long)i << 4);
  float s = 0.0f;
#pragma unroll
  for (int p = 0; p < 16; ++p) s += a[p];
  s *= (1.0f / 16.0f);
  hf[i] = s;
  hb[i] = f2bf(s);
  c[i]  = s;
}

// ---------------------------------------------------------------------------
// Per-step attention: scores = (h . A_flat[:,:,p])/sqrt(H), softmax over 16,
// attn[n][h] = sum_p A_flat[n][h][p] * w[p].  One block per batch row.
// Deterministic LDS tree reduction (no fp atomics).
// ---------------------------------------------------------------------------
__global__ __launch_bounds__(256) void attention_step(
    const float* __restrict__ h, const float* __restrict__ A,
    unsigned short* __restrict__ attn_bf) {
  __shared__ float red[16 * 257];  // padded to dodge bank conflicts
  __shared__ float w[16];
  const int n = blockIdx.x;
  const int tid = threadIdx.x;

  float ls[16];
#pragma unroll
  for (int p = 0; p < 16; ++p) ls[p] = 0.0f;

  for (int r = 0; r < 4; ++r) {
    int hi = tid + r * 256;
    float hval = h[(long)n * HH + hi];
    const float* af = A + (((long)n * HH + hi) << 4);
#pragma unroll
    for (int p = 0; p < 16; ++p) ls[p] += hval * af[p];
  }
#pragma unroll
  for (int p = 0; p < 16; ++p) red[p * 257 + tid] = ls[p];
  __syncthreads();

  for (int s = 128; s > 0; s >>= 1) {
    if (tid < s) {
#pragma unroll
      for (int p = 0; p < 16; ++p)
        red[p * 257 + tid] += red[p * 257 + tid + s];
    }
    __syncthreads();
  }

  if (tid == 0) {
    const float scale = 0.03125f;  // 1/sqrt(1024)
    float sc[16], mx = -3.4e38f;
#pragma unroll
    for (int p = 0; p < 16; ++p) {
      sc[p] = red[p * 257] * scale;
      mx = fmaxf(mx, sc[p]);
    }
    float sum = 0.0f;
#pragma unroll
    for (int p = 0; p < 16; ++p) {
      float e = __expf(sc[p] - mx);
      w[p] = e;
      sum += e;
    }
    float inv = 1.0f / sum;
#pragma unroll
    for (int p = 0; p < 16; ++p) w[p] *= inv;
  }
  __syncthreads();

  for (int r = 0; r < 4; ++r) {
    int hi = tid + r * 256;
    const float* af = A + (((long)n * HH + hi) << 4);
    float acc = 0.0f;
#pragma unroll
    for (int p = 0; p < 16; ++p) acc += af[p] * w[p];
    attn_bf[(long)n * HH + hi] = f2bf(acc);
  }
}

// ---------------------------------------------------------------------------
// Fused gate GEMM + LSTM cell.
// a = [xt | h | attn](256x3072) @ WT^T(3072x4096) + b, then per-element:
//   c' = sig(f)*c + sig(i)*tanh(g);  h' = sig(o)*tanh(c')
//
// Grid: (8 m-blocks of 32 rows) x (8 col-blocks of 128 h-columns).
// Each of the 8 waves owns a distinct 16-column group and computes
// 2 M-subtiles x 4 gates = 8 accumulators, so every B fragment is reused
// for two WMMAs and no two waves in a block load the same B data.
// The K loop is split into three branch-free 1024-wide segment loops
// (xt | h | attn), fully unrolled over segments.
// ---------------------------------------------------------------------------
__global__ __launch_bounds__(256) void lstm_gate_gemm(
    const unsigned short* __restrict__ xbf,   // [N][T][D] bf16
    const unsigned short* __restrict__ hbf,   // [N][H]    bf16 (prev h)
    const unsigned short* __restrict__ attn,  // [N][H]    bf16
    const unsigned short* __restrict__ WT,    // [4096][3072] bf16
    const float* __restrict__ bias,           // [4096] f32
    float* __restrict__ c,                    // [N][H] f32 in/out
    float* __restrict__ hnext_f,              // [N][H] f32
    unsigned short* __restrict__ hnext_bf,    // [N][H] bf16
    float* __restrict__ out,                  // [N][T][H] f32
    int t) {
  const int wave = threadIdx.x >> 5;
  const int lane = threadIdx.x & 31;
  const int m0 = blockIdx.x * 32;                 // 32-row block
  const int hcol0 = blockIdx.y * 128 + wave * 16; // 16 cols per wave
  const int l16 = lane & 15;
  const int hiHalf = lane >> 4;  // 0 or 1

  // A fragment: lane holds row (lane&15); K pattern 0..7,16..23 (lo half)
  // or 8..15,24..31 (hi half)  -> two contiguous 16B loads.
  const int a_koff = hiHalf * 8;
  // B fragment: lane holds column (lane&15); same K pattern as A
  // (0..7,16..23 lo half, 8..15,24..31 hi half) -> two 16B loads.
  const int b_koff = hiHalf * 8;

  const int nrow0 = m0 + l16;        // A row, M-subtile 0
  const int nrow1 = m0 + 16 + l16;   // A row, M-subtile 1

  // Segment base pointers for the logical concat [xt | h | attn].
  const unsigned short* aseg0[3] = {
      xbf + ((long)nrow0 * TT + t) * DD,
      hbf + (long)nrow0 * HH,
      attn + (long)nrow0 * HH};
  const unsigned short* aseg1[3] = {
      xbf + ((long)nrow1 * TT + t) * DD,
      hbf + (long)nrow1 * HH,
      attn + (long)nrow1 * HH};

  v8f acc0[4], acc1[4];
#pragma unroll
  for (int g = 0; g < 4; ++g) {
    acc0[g] = (v8f){0.f, 0.f, 0.f, 0.f, 0.f, 0.f, 0.f, 0.f};
    acc1[g] = (v8f){0.f, 0.f, 0.f, 0.f, 0.f, 0.f, 0.f, 0.f};
  }

  // Per-gate B base pointers (column = hcol0 + l16, row-major over K).
  const unsigned short* bbase[4];
#pragma unroll
  for (int g = 0; g < 4; ++g)
    bbase[g] = WT + (long)(g * HH + hcol0 + l16) * KK + b_koff;

#pragma unroll
  for (int seg = 0; seg < 3; ++seg) {
    const unsigned short* const a0 = aseg0[seg] + a_koff;
    const unsigned short* const a1 = aseg1[seg] + a_koff;
    const int kb = seg * 1024;
#pragma unroll 2
    for (int k = 0; k < 1024; k += 32) {
      FragBF fa0, fa1;
      fa0.h[0] = *reinterpret_cast<const v8us*>(a0 + k);
      fa0.h[1] = *reinterpret_cast<const v8us*>(a0 + k + 16);
      fa1.h[0] = *reinterpret_cast<const v8us*>(a1 + k);
      fa1.h[1] = *reinterpret_cast<const v8us*>(a1 + k + 16);
#pragma unroll
      for (int g = 0; g < 4; ++g) {
        const unsigned short* bp = bbase[g] + kb + k;
        FragBF fb;
        fb.h[0] = *reinterpret_cast<const v8us*>(bp);
        fb.h[1] = *reinterpret_cast<const v8us*>(bp + 16);
        acc0[g] = __builtin_amdgcn_wmma_f32_16x16x32_bf16(
            false, fa0.v, false, fb.v, (short)0, acc0[g], false, false);
        acc1[g] = __builtin_amdgcn_wmma_f32_16x16x32_bf16(
            false, fa1.v, false, fb.v, (short)0, acc1[g], false, false);
      }
    }
  }

  // Epilogue: C/D layout -> row = e + 8*hiHalf, col = lane&15.
  const int hcol = hcol0 + l16;
  const float bi = bias[hcol];
  const float bf = bias[HH + hcol];
  const float bo = bias[2 * HH + hcol];
  const float bg = bias[3 * HH + hcol];
#pragma unroll
  for (int sub = 0; sub < 2; ++sub) {
    const v8f* ai = sub ? acc1 : acc0;
#pragma unroll
    for (int e = 0; e < 8; ++e) {
      const int n = m0 + sub * 16 + e + hiHalf * 8;
      const long idx = (long)n * HH + hcol;
      float iv = ai[0][e] + bi;
      float fv = ai[1][e] + bf;
      float ov = ai[2][e] + bo;
      float gv = ai[3][e] + bg;
      float cn = sigf(fv) * c[idx] + sigf(iv) * tanhf(gv);
      float hn = sigf(ov) * tanhf(cn);
      c[idx] = cn;
      hnext_f[idx] = hn;
      hnext_bf[idx] = f2bf(hn);
      out[((long)n * TT + t) * HH + hcol] = hn;
    }
  }
}

// ---------------------------------------------------------------------------
extern "C" void kernel_launch(void* const* d_in, const int* in_sizes, int n_in,
                              void* d_out, int out_size, void* d_ws,
                              size_t ws_size, hipStream_t stream) {
  const float* x  = (const float*)d_in[0];   // (N,T,D)
  const float* A  = (const float*)d_in[1];   // (N,H,4,4)
  const float* Wx = (const float*)d_in[2];   // (D,4H)
  const float* Wh = (const float*)d_in[3];   // (H,4H)
  const float* Wa = (const float*)d_in[4];   // (H,4H)
  const float* b  = (const float*)d_in[5];   // (4H,)
  float* out = (float*)d_out;                // (N,T,H)

  char* ws = (char*)d_ws;
  unsigned short* WT = (unsigned short*)ws;  ws += (size_t)NC * KK * 2;      // 24 MB
  unsigned short* xb = (unsigned short*)ws;  ws += (size_t)NB * TT * DD * 2; // 32 MB
  float* hf[2];
  hf[0] = (float*)ws;  ws += (size_t)NB * HH * 4;
  hf[1] = (float*)ws;  ws += (size_t)NB * HH * 4;
  unsigned short* hb[2];
  hb[0] = (unsigned short*)ws;  ws += (size_t)NB * HH * 2;
  hb[1] = (unsigned short*)ws;  ws += (size_t)NB * HH * 2;
  float* c = (float*)ws;  ws += (size_t)NB * HH * 4;
  unsigned short* attn = (unsigned short*)ws;  ws += (size_t)NB * HH * 2;

  {
    long tot = (long)NC * KK;
    convert_weights<<<dim3((unsigned)((tot + 255) / 256)), 256, 0, stream>>>(
        Wx, Wh, Wa, WT);
  }
  {
    long tot = (long)NB * TT * DD;
    convert_f32_bf16<<<dim3((unsigned)((tot + 255) / 256)), 256, 0, stream>>>(
        x, xb, tot);
  }
  init_state<<<dim3((NB * HH) / 256), 256, 0, stream>>>(A, hf[0], hb[0], c);

  int cur = 0;
  for (int t = 0; t < TT; ++t) {
    attention_step<<<dim3(NB), 256, 0, stream>>>(hf[cur], A, attn);
    lstm_gate_gemm<<<dim3(NB / 32, HH / 128), 256, 0, stream>>>(
        xb, hb[cur], attn, WT, b, c, hf[1 - cur], hb[1 - cur], out, t);
    cur ^= 1;
  }
}
